// DynamicConnectogramAttention_70884140253738
// MI455X (gfx1250) — matrix operand order repaired and checked by reading, not compiled, not run on hardware
//
#include <hip/hip_runtime.h>
#include <math.h>

// ---- problem constants (from reference) ----
#define B_   32
#define F_   16
#define N_   64
#define T_   1024
#define D_   32
#define H_   8
#define EH_  128     // T/H
#define KSP_ 32      // top-k
#define P1_  8

typedef __attribute__((ext_vector_type(16))) __bf16 v16bf;
typedef __attribute__((ext_vector_type(8)))  __bf16 v8bf;
typedef __attribute__((ext_vector_type(8)))  float  v8f;

// native bf16 conversions (gfx1250 has hardware BF16 cvt)
__device__ __forceinline__ __bf16 f2bf(float x)  { return (__bf16)x; }
__device__ __forceinline__ float  bf2f(__bf16 h) { return (float)h; }

__device__ __forceinline__ v8f wmma_bf16(v16bf a, v16bf b, v8f c) {
  // D = A(16x32) * B(32x16) + C, f32 accumulate
  return __builtin_amdgcn_wmma_f32_16x16x32_bf16(false, a, false, b, (short)0, c,
                                                 false, false);
}

// Build an A-fragment from a row-major bf16 row: elements [k0..k0+7] and
// [k0+8..k0+15] (contiguous k order, matching the B fragments).
// Both chunks are 16B-aligned contiguous LDS reads -> 2x ds_load_b128.
__device__ __forceinline__ v16bf load_a_frag(const __bf16* row, int k0) {
  v8bf lo = *reinterpret_cast<const v8bf*>(row + k0);
  v8bf hi = *reinterpret_cast<const v8bf*>(row + k0 + 8);
  return __builtin_shufflevector(lo, hi, 0, 1, 2, 3, 4, 5, 6, 7,
                                         8, 9, 10, 11, 12, 13, 14, 15);
}

// ---------------------------------------------------------------------------
// Kernel 0: zero the stats accumulators in workspace.
// stats layout (floats): [0,F) bn1 sum | [F,2F) bn1 sumsq | [2F,3F) bn2 sum | [3F,4F) bn2 sumsq
// ---------------------------------------------------------------------------
__global__ void k_init_stats(float* stats) {
  int i = threadIdx.x;
  if (i < 4 * F_) stats[i] = 0.f;
}

// ---------------------------------------------------------------------------
// Kernel 1: BN1 batch statistics of the stem conv output (per channel f).
// ---------------------------------------------------------------------------
__global__ __launch_bounds__(256) void k_stem_stats(const float* __restrict__ x,
                                                    const float* __restrict__ conv_w,
                                                    float* __restrict__ stats) {
  __shared__ float rsum[256], rsq[256];
  int b = blockIdx.x / F_;
  int f = blockIdx.x % F_;
  float w0 = conv_w[f*3+0], w1 = conv_w[f*3+1], w2 = conv_w[f*3+2];
  const float* xb = x + (size_t)b * N_ * T_;
  float s = 0.f, q = 0.f;
  for (int idx = threadIdx.x; idx < N_ * T_; idx += 256) {
    int n = idx >> 10, t = idx & (T_ - 1);
    const float* xr = xb + n * T_;
    float xm = (t > 0)      ? xr[t-1] : 0.f;
    float xc =                xr[t];
    float xp = (t < T_-1)   ? xr[t+1] : 0.f;
    float c = w0*xm + w1*xc + w2*xp;
    s += c; q += c * c;
  }
  rsum[threadIdx.x] = s; rsq[threadIdx.x] = q;
  __syncthreads();
  for (int st = 128; st > 0; st >>= 1) {
    if (threadIdx.x < st) {
      rsum[threadIdx.x] += rsum[threadIdx.x + st];
      rsq[threadIdx.x]  += rsq[threadIdx.x + st];
    }
    __syncthreads();
  }
  if (threadIdx.x == 0) {
    atomicAdd(&stats[f],      rsum[0]);
    atomicAdd(&stats[F_ + f], rsq[0]);
  }
}

// ---------------------------------------------------------------------------
// Kernel 2: fully fused per-(b,f,head) attention (see round-1 analysis).
// ---------------------------------------------------------------------------
__global__ __launch_bounds__(256) void k_attention(
    const float* __restrict__ x,   const float* __restrict__ conv_w,
    const float* __restrict__ bn1_g, const float* __restrict__ bn1_b,
    const float* __restrict__ q_w,   const float* __restrict__ q_b,
    const float* __restrict__ kw0, const float* __restrict__ kw1, const float* __restrict__ kw2,
    const float* __restrict__ kb0, const float* __restrict__ kb1, const float* __restrict__ kb2,
    const float* __restrict__ vw0, const float* __restrict__ vw1, const float* __restrict__ vw2,
    const float* __restrict__ vb0, const float* __restrict__ vb1, const float* __restrict__ vb2,
    float* __restrict__ stats, float* __restrict__ out)
{
  __shared__ __align__(32) __bf16 hs  [N_][132];   // h tile, t = t0-2 .. t0+129
  __shared__ __align__(32) __bf16 hTs [132][N_];   // transposed (B-operand for Q)
  __shared__ __align__(32) __bf16 ks_ [N_][EH_];   // K tile  (B-operand for S)
  __shared__ __align__(32) __bf16 vTs [EH_][N_];   // V transposed (B-operand for G)
  __shared__ __align__(32) float  qf  [D_][EH_];   // Q f32 (residual)
  __shared__ __align__(32) __bf16 qbf [D_][EH_];   // Q bf16 (A-operand for S)
  __shared__               float  sc  [D_][N_];    // scores / exp values
  __shared__ __align__(32) __bf16 adjbf[D_][N_];   // softmax adj (A-operand for G)
  __shared__               float  gbuf[D_][EH_];   // gelu output
  __shared__ float rowred[D_][8];
  __shared__ float thrsh[D_], rmaxs[D_], denoms[D_];
  __shared__ float red[256], redq[256];

  const int tid  = threadIdx.x;
  const int lane = tid & 31;
  const int wave = tid >> 5;
  const int hi   = lane >> 4;           // wave32 half
  const int bid  = blockIdx.x;
  const int b    = bid / (F_ * H_);
  const int f    = (bid / H_) % F_;
  const int head = bid % H_;
  const int t0   = head * EH_;

  // BN1 parameters from pre-pass stats (eps = 0)
  const float cnt1 = (float)B_ * N_ * T_;
  float mu1  = stats[f] / cnt1;
  float var1 = stats[F_ + f] / cnt1 - mu1 * mu1;
  float rs1  = rsqrtf(var1);
  float g1 = bn1_g[f], be1 = bn1_b[f];
  float cw0 = conv_w[f*3+0], cw1v = conv_w[f*3+1], cw2v = conv_w[f*3+2];
  const float* xb = x + (size_t)b * N_ * T_;

  // ---- Phase 1: h tile with +-2 halo (zero outside [0,T)) ----
  for (int idx = tid; idx < N_ * 132; idx += 256) {
    int n = idx / 132, tc = idx % 132;
    int t = t0 - 2 + tc;
    float val = 0.f;
    if (t >= 0 && t < T_) {
      const float* xr = xb + n * T_;
      float xm = (t > 0)    ? xr[t-1] : 0.f;
      float xc =              xr[t];
      float xp = (t < T_-1) ? xr[t+1] : 0.f;
      val = g1 * ((cw0*xm + cw1v*xc + cw2v*xp) - mu1) * rs1 + be1;
    }
    __bf16 hv = f2bf(val);
    hs[n][tc]  = hv;
    hTs[tc][n] = hv;
  }
  __syncthreads();

  // ---- Phase 2: inception K (row-major) and V (transposed) ----
  float ka = kw0[f];
  float k1a = kw1[f*3+0], k1b = kw1[f*3+1], k1c = kw1[f*3+2];
  float k2a = kw2[f*5+0], k2b = kw2[f*5+1], k2c = kw2[f*5+2], k2d = kw2[f*5+3], k2e = kw2[f*5+4];
  float kbias = kb0[f] + kb1[f] + kb2[f];
  float va = vw0[f];
  float v1a = vw1[f*3+0], v1b = vw1[f*3+1], v1c = vw1[f*3+2];
  float v2a = vw2[f*5+0], v2b = vw2[f*5+1], v2c = vw2[f*5+2], v2d = vw2[f*5+3], v2e = vw2[f*5+4];
  float vbias = vb0[f] + vb1[f] + vb2[f];
  const float third = 1.f / 3.f;
  for (int idx = tid; idx < N_ * EH_; idx += 256) {
    int n = idx >> 7, e = idx & (EH_ - 1);
    float h0 = bf2f(hs[n][e+0]), h1 = bf2f(hs[n][e+1]), h2 = bf2f(hs[n][e+2]);
    float h3 = bf2f(hs[n][e+3]), h4 = bf2f(hs[n][e+4]);
    float kv = (ka*h2 + k1a*h1 + k1b*h2 + k1c*h3
              + k2a*h0 + k2b*h1 + k2c*h2 + k2d*h3 + k2e*h4 + kbias) * third;
    float vv = (va*h2 + v1a*h1 + v1b*h2 + v1c*h3
              + v2a*h0 + v2b*h1 + v2c*h2 + v2d*h3 + v2e*h4 + vbias) * third;
    ks_[n][e] = f2bf(kv);
    vTs[e][n] = f2bf(vv);
  }
  __syncthreads();

  // ---- Phase 3: Q = Wq(32x64) x h(64x128)  [WMMA], wave w owns 16 t-columns ----
  {
    int ecol = wave * 16 + (lane & 15);
    v16bf aW[2][2];
    #pragma unroll
    for (int mt = 0; mt < 2; ++mt) {
      int m = mt*16 + (lane & 15);
      const float* wr = q_w + (size_t)(f*D_ + m) * N_;
      #pragma unroll
      for (int kt = 0; kt < 2; ++kt) {
        int k0 = kt*32 + hi*16;
        v16bf a;
        #pragma unroll
        for (int i = 0; i < 8; ++i) { a[i] = f2bf(wr[k0+i]); a[8+i] = f2bf(wr[k0+8+i]); }
        aW[mt][kt] = a;
      }
    }
    #pragma unroll
    for (int mt = 0; mt < 2; ++mt) {
      v8f acc = {};
      #pragma unroll
      for (int kt = 0; kt < 2; ++kt) {
        v16bf bfr = *reinterpret_cast<const v16bf*>(&hTs[ecol + 2][kt*32 + hi*16]);
        acc = wmma_bf16(aW[mt][kt], bfr, acc);
      }
      #pragma unroll
      for (int r = 0; r < 8; ++r) {
        int m = mt*16 + hi*8 + r;
        float qv = acc[r] + q_b[f*D_ + m];
        qf[m][ecol]  = qv;
        qbf[m][ecol] = f2bf(qv);
      }
    }
  }
  __syncthreads();

  // ---- Phase 4: S = Q x K^T * (1/sqrt(T))  [WMMA], 8 waves = 2x4 tiles ----
  {
    int mt = wave >> 2, nt = wave & 3;
    int ncol = nt*16 + (lane & 15);
    int m    = mt*16 + (lane & 15);
    v8f acc = {};
    #pragma unroll
    for (int kt = 0; kt < 4; ++kt) {
      v16bf a   = load_a_frag(&qbf[m][0], kt*32 + hi*16);
      v16bf bfr = *reinterpret_cast<const v16bf*>(&ks_[ncol][kt*32 + hi*16]);
      acc = wmma_bf16(a, bfr, acc);
    }
    #pragma unroll
    for (int r = 0; r < 8; ++r) {
      int mr = mt*16 + hi*8 + r;
      sc[mr][ncol] = acc[r] * 0.03125f;   // 1/sqrt(1024)
    }
  }
  __syncthreads();

  // ---- Phase 5: per-row top-k(32) threshold + softmax (32 rows x 8 threads) ----
  {
    int m = tid >> 3, g = tid & 7;
    float lmin = 3.4e38f, lmax = -3.4e38f;
    #pragma unroll
    for (int j = 0; j < 8; ++j) {
      int n = g*8 + j;
      float xv = sc[m][n];
      lmax = fmaxf(lmax, xv);
      int cnt = 0;
      for (int nn = 0; nn < N_; ++nn) cnt += (sc[m][nn] > xv) ? 1 : 0;
      if (cnt < KSP_) lmin = fminf(lmin, xv);   // candidate k-th largest
    }
    rowred[m][g] = lmin;
    __syncthreads();
    if (g == 0) { float t = rowred[m][0];
      for (int j = 1; j < 8; ++j) t = fminf(t, rowred[m][j]); thrsh[m] = t; }
    __syncthreads();
    rowred[m][g] = lmax;
    __syncthreads();
    if (g == 0) { float t = rowred[m][0];
      for (int j = 1; j < 8; ++j) t = fmaxf(t, rowred[m][j]); rmaxs[m] = t; }
    __syncthreads();
    float thr = thrsh[m], rmx = rmaxs[m], psum = 0.f;
    #pragma unroll
    for (int j = 0; j < 8; ++j) {
      int n = g*8 + j;
      float xv = sc[m][n];
      float ev = (xv >= thr) ? __expf(xv - rmx) : 0.f;
      sc[m][n] = ev;
      psum += ev;
    }
    rowred[m][g] = psum;
    __syncthreads();
    if (g == 0) { float t = 0.f;
      for (int j = 0; j < 8; ++j) t += rowred[m][j]; denoms[m] = t; }
    __syncthreads();
    float inv = 1.f / denoms[m];
    #pragma unroll
    for (int j = 0; j < 8; ++j) {
      int n = g*8 + j;
      adjbf[m][n] = f2bf(sc[m][n] * inv);
    }
  }
  __syncthreads();

  // ---- Phase 6: G = adj x V + Q, exact GELU  [WMMA], 16 tiles / 8 waves ----
  {
    int mt = wave >> 2;
    #pragma unroll
    for (int qi = 0; qi < 2; ++qi) {
      int et = (wave & 3)*2 + qi;
      int ecol = et*16 + (lane & 15);
      int m    = mt*16 + (lane & 15);
      v8f acc = {};
      #pragma unroll
      for (int kt = 0; kt < 2; ++kt) {
        v16bf a   = load_a_frag(&adjbf[m][0], kt*32 + hi*16);
        v16bf bfr = *reinterpret_cast<const v16bf*>(&vTs[ecol][kt*32 + hi*16]);
        acc = wmma_bf16(a, bfr, acc);
      }
      #pragma unroll
      for (int r = 0; r < 8; ++r) {
        int mr = mt*16 + hi*8 + r;
        float gv = acc[r] + qf[mr][ecol];                      // residual +Q
        gbuf[mr][ecol] = 0.5f * gv * (1.f + erff(gv * 0.70710678118f));
      }
    }
  }
  __syncthreads();

  // ---- Phase 7: BN2 stats atomics + avg-pool(8) -> out (pre-affine) ----
  {
    float s = 0.f, q2 = 0.f;
    for (int idx = tid; idx < D_ * EH_; idx += 256) {
      float v = gbuf[idx >> 7][idx & 127];
      s += v; q2 += v * v;
    }
    red[tid] = s; redq[tid] = q2;
    __syncthreads();
    for (int st = 128; st > 0; st >>= 1) {
      if (tid < st) { red[tid] += red[tid+st]; redq[tid] += redq[tid+st]; }
      __syncthreads();
    }
    if (tid == 0) {
      atomicAdd(&stats[2*F_ + f], red[0]);
      atomicAdd(&stats[3*F_ + f], redq[0]);
    }
    for (int idx = tid; idx < D_ * 16; idx += 256) {
      int m = idx >> 4, pe = idx & 15;
      float sacc = 0.f;
      #pragma unroll
      for (int j = 0; j < 8; ++j) sacc += gbuf[m][pe*8 + j];
      out[(((size_t)b * F_ + f) * D_ + m) * 128 + head*16 + pe] = sacc * 0.125f;
    }
  }
}

// ---------------------------------------------------------------------------
// Kernel 3: apply BN2 affine in place (pooling commutes with per-channel affine).
// ---------------------------------------------------------------------------
__global__ __launch_bounds__(256) void k_finalize(float* __restrict__ out,
                                                  const float* __restrict__ stats,
                                                  const float* __restrict__ bn2_g,
                                                  const float* __restrict__ bn2_b) {
  int idx = blockIdx.x * 256 + threadIdx.x;
  if (idx >= B_ * F_ * D_ * 128) return;
  int f = (idx >> 12) & (F_ - 1);          // D*128 = 4096 per (b,f) channel-d slab
  const float cnt2 = (float)B_ * D_ * T_;
  float mu  = stats[2*F_ + f] / cnt2;
  float var = stats[3*F_ + f] / cnt2 - mu * mu;
  float rs  = rsqrtf(var + 1e-5f);
  out[idx] = bn2_g[f] * (out[idx] - mu) * rs + bn2_b[f];
}

// ---------------------------------------------------------------------------
extern "C" void kernel_launch(void* const* d_in, const int* in_sizes, int n_in,
                              void* d_out, int out_size, void* d_ws, size_t ws_size,
                              hipStream_t stream) {
  (void)in_sizes; (void)n_in; (void)out_size; (void)ws_size;
  const float* x      = (const float*)d_in[0];
  const float* conv_w = (const float*)d_in[1];
  const float* bn1_g  = (const float*)d_in[2];
  const float* bn1_b  = (const float*)d_in[3];
  const float* q_w    = (const float*)d_in[4];
  const float* q_b    = (const float*)d_in[5];
  const float* kw0    = (const float*)d_in[6];
  const float* kw1    = (const float*)d_in[7];
  const float* kw2    = (const float*)d_in[8];
  const float* kb0    = (const float*)d_in[9];
  const float* kb1    = (const float*)d_in[10];
  const float* kb2    = (const float*)d_in[11];
  const float* vw0    = (const float*)d_in[12];
  const float* vw1    = (const float*)d_in[13];
  const float* vw2    = (const float*)d_in[14];
  const float* vb0    = (const float*)d_in[15];
  const float* vb1    = (const float*)d_in[16];
  const float* vb2    = (const float*)d_in[17];
  const float* bn2_g  = (const float*)d_in[18];
  const float* bn2_b  = (const float*)d_in[19];
  float* out   = (float*)d_out;
  float* stats = (float*)d_ws;

  hipLaunchKernelGGL(k_init_stats, dim3(1), dim3(64), 0, stream, stats);
  hipLaunchKernelGGL(k_stem_stats, dim3(B_ * F_), dim3(256), 0, stream, x, conv_w, stats);
  hipLaunchKernelGGL(k_attention, dim3(B_ * F_ * H_), dim3(256), 0, stream,
                     x, conv_w, bn1_g, bn1_b, q_w, q_b,
                     kw0, kw1, kw2, kb0, kb1, kb2,
                     vw0, vw1, vw2, vb0, vb1, vb2, stats, out);
  hipLaunchKernelGGL(k_finalize, dim3((B_ * F_ * D_ * 128 + 255) / 256), dim3(256),
                     0, stream, out, stats, bn2_g, bn2_b);
}
